// Flow_63883343560917
// MI455X (gfx1250) — matrix operand order repaired and checked by reading, not compiled, not run on hardware
//
#include <hip/hip_runtime.h>
#include <hip/hip_bf16.h>
#include <stdint.h>

typedef __attribute__((ext_vector_type(16))) _Float16 v16h;
typedef __attribute__((ext_vector_type(8)))  float    v8f;

#define N_TRI          200000
#define ROWS_PER_BLOCK 128
#define ACT_STRIDE     136        // f16 elems per LDS row (128 + 8 pad); 272B = 68 banks -> conflict-free
#define ACT_V4         17         // uint4 (16B) per row stride = 136/8
#define STOPPED        (-100.0f)

// ---- workspace layout (dword offsets) ----
// hdr: [0..2]=v [3..5]=w [6..8]=u [9]=invScale [10..12]=recv [15]=lastObject(int)
#define WS_HDR    0
#define WS_BH0    16                        // 64 floats: folded head bias
#define WS_W0PACK 80                        // 8  frags * 256 dwords
#define WS_W1PACK (WS_W0PACK + 8*256)       // 32 frags
#define WS_W2PACK (WS_W1PACK + 32*256)      // 16 frags
#define WS_H0PACK (WS_W2PACK + 16*256)      // 8  frags
#define WS_H1PACK (WS_H0PACK + 8*256)       // 8  frags

// =====================================================================
// Prep 1: basis/scale, scene MLP on the <=6 pc rows, fold pc_emb into
// head-layer-0 bias, last_object.  One block of 128 threads; trivial cost.
// =====================================================================
__global__ __launch_bounds__(128)
void prep_scalar(const float* __restrict__ tv, const float* __restrict__ tm,
                 const float* __restrict__ rc, const int* __restrict__ pc,
                 const float* __restrict__ sW0, const float* __restrict__ sb0,
                 const float* __restrict__ sW1, const float* __restrict__ sb1,
                 const float* __restrict__ sW2, const float* __restrict__ sb2,
                 const float* __restrict__ hW0, const float* __restrict__ hb0,
                 float* __restrict__ ws)
{
    __shared__ float basis[16];
    __shared__ float xyz6[6][12];
    __shared__ float h1[6][128];
    __shared__ float h2b[6][128];
    __shared__ float obj6[6][64];
    __shared__ int   pcs[6];

    int t = threadIdx.x;
    if (t < 6) pcs[t] = pc[t];
    if (t == 0) {
        float d0 = tm[0]-rc[0], d1 = tm[1]-rc[1], d2 = tm[2]-rc[2];
        float sc = sqrtf(d0*d0 + d1*d1 + d2*d2);
        float u0 = d0/sc, u1 = d1/sc, u2 = d2/sc;
        float a0 = fabsf(u0), a1 = fabsf(u1), a2 = fabsf(u2);
        float e0 = 0.f, e1 = 0.f, e2 = 0.f;            // first-min semantics of argmin
        if (a0 <= a1 && a0 <= a2) e0 = 1.f; else if (a1 <= a2) e1 = 1.f; else e2 = 1.f;
        float v0 = u1*e2 - u2*e1, v1 = u2*e0 - u0*e2, v2 = u0*e1 - u1*e0;
        float vn = sqrtf(v0*v0 + v1*v1 + v2*v2);
        v0 /= vn; v1 /= vn; v2 /= vn;
        float w0 = u1*v2 - u2*v1, w1 = u2*v0 - u0*v2, w2 = u0*v1 - u1*v0;
        basis[0]=v0; basis[1]=v1; basis[2]=v2;
        basis[3]=w0; basis[4]=w1; basis[5]=w2;
        basis[6]=u0; basis[7]=u1; basis[8]=u2;
        basis[9]=1.0f/sc;
        basis[10]=rc[0]; basis[11]=rc[1]; basis[12]=rc[2];
        for (int q = 0; q < 13; ++q) ws[WS_HDR + q] = basis[q];
        // last_object
        int i = -1;
        for (int q = 5; q >= 0; --q) if (pc[q] == -1) i = q;   // first hit
        int j = i - 1;
        int last = (i >= 0 && j >= 0) ? pc[j] : -1;
        ((int*)ws)[15] = last;
    }
    __syncthreads();

    if (t < 6) {                     // transformed coords for pc rows (clipped)
        int r = pcs[t] >= 0 ? pcs[t] : 0;
        const float* p = tv + (size_t)r * 9;
        float iv = basis[9];
        for (int vtx = 0; vtx < 3; ++vtx) {
            float q0 = (p[vtx*3+0]-basis[10])*iv;
            float q1 = (p[vtx*3+1]-basis[11])*iv;
            float q2 = (p[vtx*3+2]-basis[12])*iv;
            xyz6[t][vtx*3+0] = q0*basis[0] + q1*basis[1] + q2*basis[2];
            xyz6[t][vtx*3+1] = q0*basis[3] + q1*basis[4] + q2*basis[5];
            xyz6[t][vtx*3+2] = q0*basis[6] + q1*basis[7] + q2*basis[8];
        }
    }
    __syncthreads();
    if (t < 128)
        for (int o = 0; o < 6; ++o) {
            float s = sb0[t];
            for (int k = 0; k < 9; ++k) s += xyz6[o][k] * sW0[t*9 + k];
            h1[o][t] = fmaxf(s, 0.0f);
        }
    __syncthreads();
    if (t < 128)
        for (int o = 0; o < 6; ++o) {
            float s = sb1[t];
            for (int k = 0; k < 128; ++k) s += h1[o][k] * sW1[t*128 + k];
            h2b[o][t] = fmaxf(s, 0.0f);
        }
    __syncthreads();
    if (t < 64)
        for (int o = 0; o < 6; ++o) {
            float s = sb2[t];
            for (int k = 0; k < 128; ++k) s += h2b[o][k] * sW2[t*128 + k];
            obj6[o][t] = s;
        }
    __syncthreads();
    if (t < 64) {                    // fold constant pc_emb through head W0
        float s = hb0[t];
        for (int o = 0; o < 6; ++o)
            if (pcs[o] >= 0)
                for (int k = 0; k < 64; ++k)
                    s += obj6[o][k] * hW0[t*448 + 64 + o*64 + k];
        ws[WS_BH0 + t] = s;
    }
}

// =====================================================================
// Prep 2: pack all weights into WMMA B-fragment order (f16).
// Fragment = 16x32 tile of W^T for one (kt,nt).  Lane L holds column
// n = nt*16+(L&15); VGPR v holds K pair k = kt*32 + 2(v&3) + (v>=4 ? 16:0) + (L>=16 ? 8:0).
// Stored contiguously (32 lanes * 8 dwords) so main-kernel loads coalesce.
// =====================================================================
__global__ __launch_bounds__(32)
void prep_pack(const float* __restrict__ sW0, const float* __restrict__ sW1,
               const float* __restrict__ sW2, const float* __restrict__ hW0,
               const float* __restrict__ hW1, float* __restrict__ ws)
{
    int f    = blockIdx.x;      // 0..71 global fragment id
    int lane = threadIdx.x;     // 0..31
    uint32_t* wsu = (uint32_t*)ws;
    const float* W; int ldw, kreal, ntiles, lf; uint32_t* dst;
    if (f < 8)       { W = sW0; ldw = 9;   kreal = 9;   ntiles = 8; lf = f;      dst = wsu + WS_W0PACK; }
    else if (f < 40) { W = sW1; ldw = 128; kreal = 128; ntiles = 8; lf = f - 8;  dst = wsu + WS_W1PACK; }
    else if (f < 56) { W = sW2; ldw = 128; kreal = 128; ntiles = 4; lf = f - 40; dst = wsu + WS_W2PACK; }
    else if (f < 64) { W = hW0; ldw = 448; kreal = 64;  ntiles = 4; lf = f - 56; dst = wsu + WS_H0PACK; }
    else             { W = hW1; ldw = 64;  kreal = 64;  ntiles = 4; lf = f - 64; dst = wsu + WS_H1PACK; }

    int kt = lf / ntiles, nt = lf % ntiles;
    int n  = nt*16 + (lane & 15);
    int kh = (lane >= 16) ? 8 : 0;
    uint32_t out[8];
    #pragma unroll
    for (int v = 0; v < 8; ++v) {
        int k = kt*32 + 2*(v & 3) + ((v & 4) ? 16 : 0) + kh;
        float w0 = (k     < kreal) ? W[n*ldw + k]     : 0.0f;
        float w1 = (k + 1 < kreal) ? W[n*ldw + k + 1] : 0.0f;
        union { _Float16 h[2]; uint32_t u; } pk;
        pk.h[0] = (_Float16)w0; pk.h[1] = (_Float16)w1;
        out[v] = pk.u;
    }
    uint32_t* d = dst + ((size_t)lf*32 + lane)*8;
    #pragma unroll
    for (int v = 0; v < 8; ++v) d[v] = out[v];
}

// =====================================================================
// Main kernel: 8 waves * 16-row strips = 128 rows/block.
// Each wave reads/writes only its own LDS rows -> no workgroup barriers.
// =====================================================================
union FragH { v16h h; uint32_t u[8]; uint4 q[2]; };

// GEMM layer: act rows m0..m0+15, cols [0,KT*32) (f16, LDS)  @  packed W^T
//   -> act cols [0,NT*16).   ACTMODE: 0=linear, 1=relu, 2=leaky(0.01)
template<int KT, int NT, int ACTMODE>
__device__ __forceinline__ void gemm_layer(_Float16* act, const uint4* actV,
                                           const uint32_t* __restrict__ pack,
                                           const float* __restrict__ bias,
                                           int m0, int lane)
{
    const int mh = lane & 15;
    const int hi = (lane >= 16) ? 1 : 0;
    const int m  = m0 + mh;

    // A fragments per ISA 16-bit A layout: VGPR0-3 = K {0..7}+hi*8 (16B),
    // VGPR4-7 = K {16..23}+hi*8 (16B)  -> two ds_load_b128 per fragment.
    FragH a[KT];
    #pragma unroll
    for (int kt = 0; kt < KT; ++kt) {
        a[kt].q[0] = actV[m*ACT_V4 + kt*4 + hi];
        a[kt].q[1] = actV[m*ACT_V4 + kt*4 + hi + 2];
    }
    asm volatile("" ::: "memory");             // A loads precede in-place stores

    const int nloc = lane & 15;
    #pragma unroll
    for (int nt = 0; nt < NT; ++nt) {
        v8f c = {0.f,0.f,0.f,0.f,0.f,0.f,0.f,0.f};
        #pragma unroll
        for (int kt = 0; kt < KT; ++kt) {
            FragH b;
            const uint32_t* q = pack + ((size_t)(kt*NT + nt)*32 + lane)*8;
            #pragma unroll
            for (int v = 0; v < 8; ++v) b.u[v] = q[v];
            c = __builtin_amdgcn_wmma_f32_16x16x32_f16(false, a[kt].h, false, b.h,
                                                       (short)0, c, false, false);
        }
        float bv = bias[nt*16 + nloc];
        #pragma unroll
        for (int r = 0; r < 8; ++r) {          // D layout: (M=m0+r+hi*8, N=nt*16+nloc)
            float x = c[r] + bv;
            if (ACTMODE == 1)      x = fmaxf(x, 0.0f);
            else if (ACTMODE == 2) x = (x > 0.0f) ? x : 0.01f * x;
            act[(m0 + r + hi*8)*ACT_STRIDE + nt*16 + nloc] = (_Float16)x;
        }
    }
    asm volatile("" ::: "memory");
}

__global__ __launch_bounds__(256)
void flow_main(const float* __restrict__ tv, const unsigned char* __restrict__ mask,
               const float* __restrict__ sb0, const float* __restrict__ sb1,
               const float* __restrict__ sb2, const float* __restrict__ hb1,
               const float* __restrict__ hW2, const float* __restrict__ hb2,
               const float* __restrict__ ws, float* __restrict__ out)
{
    __shared__ _Float16 act[ROWS_PER_BLOCK * ACT_STRIDE];   // ~34.8 KB
    uint4* actV = (uint4*)act;
    const uint32_t* wsu = (const uint32_t*)ws;

    const int tid  = threadIdx.x;
    const int lane = tid & 31;
    const int wave = tid >> 5;
    const int m0   = wave * 16;
    const int rowBase = blockIdx.x * ROWS_PER_BLOCK;

    float bss[13];
    #pragma unroll
    for (int i = 0; i < 13; ++i) bss[i] = ws[i];
    const int lastObj = ((const int*)ws)[15];

    // ---- geometry: lanes<16 each fill their own row: 32 f16 cols as 4x b128 ----
    if (lane < 16) {
        int m = m0 + lane;
        int g = rowBase + m;
        uint32_t rowu[16];
        #pragma unroll
        for (int i = 0; i < 16; ++i) rowu[i] = 0u;
        if (g < N_TRI) {
            const float* p = tv + (size_t)g * 9;
            float iv = bss[9];
            _Float16 hc[10];
            #pragma unroll
            for (int vtx = 0; vtx < 3; ++vtx) {
                float q0 = (p[vtx*3+0]-bss[10])*iv;
                float q1 = (p[vtx*3+1]-bss[11])*iv;
                float q2 = (p[vtx*3+2]-bss[12])*iv;
                hc[vtx*3+0] = (_Float16)(q0*bss[0] + q1*bss[1] + q2*bss[2]);
                hc[vtx*3+1] = (_Float16)(q0*bss[3] + q1*bss[4] + q2*bss[5]);
                hc[vtx*3+2] = (_Float16)(q0*bss[6] + q1*bss[7] + q2*bss[8]);
            }
            hc[9] = (_Float16)0.0f;
            #pragma unroll
            for (int i = 0; i < 5; ++i) {
                union { _Float16 h[2]; uint32_t u; } pk;
                pk.h[0] = hc[2*i]; pk.h[1] = hc[2*i+1];
                rowu[i] = pk.u;
            }
        }
        #pragma unroll
        for (int i = 0; i < 4; ++i)
            actV[m*ACT_V4 + i] = make_uint4(rowu[4*i], rowu[4*i+1], rowu[4*i+2], rowu[4*i+3]);
    }
    asm volatile("" ::: "memory");

    // ---- MLP pipeline, all WMMA (72 per wave) ----
    gemm_layer<1, 8, 1>(act, actV, wsu + WS_W0PACK, sb0,          m0, lane);  // 9(->32) -> 128, relu
    gemm_layer<4, 8, 1>(act, actV, wsu + WS_W1PACK, sb1,          m0, lane);  // 128 -> 128, relu
    gemm_layer<4, 4, 0>(act, actV, wsu + WS_W2PACK, sb2,          m0, lane);  // 128 -> 64 (obj)
    gemm_layer<2, 4, 2>(act, actV, wsu + WS_H0PACK, ws + WS_BH0,  m0, lane);  // 64 -> 64, leaky, folded bias
    gemm_layer<2, 4, 2>(act, actV, wsu + WS_H1PACK, hb1,          m0, lane);  // 64 -> 64, leaky

    // ---- final 64 -> 1 matvec: 2 lanes per row, 4x b128 LDS loads each ----
    const int mh = lane & 15;
    const int half = lane >> 4;
    const int m = m0 + mh;
    float s = 0.0f;
    #pragma unroll
    for (int j = 0; j < 4; ++j) {
        union { uint4 q; _Float16 h[8]; } blk;
        blk.q = actV[m*ACT_V4 + half*4 + j];
        #pragma unroll
        for (int e = 0; e < 8; ++e)
            s += (float)blk.h[e] * hW2[half*32 + j*8 + e];
    }
    s += __shfl_xor(s, 16, 32);

    if (half == 0) {
        int g = rowBase + m;
        if (g < N_TRI) {
            float val = s + hb2[0];
            if (!mask[g])                       val = STOPPED;
            if (lastObj >= 0 && g == lastObj)   val = STOPPED;
            out[g] = val;
        }
    }
}

// =====================================================================
extern "C" void kernel_launch(void* const* d_in, const int* in_sizes, int n_in,
                              void* d_out, int out_size, void* d_ws, size_t ws_size,
                              hipStream_t stream)
{
    const float* tv  = (const float*)d_in[0];
    const float* tm  = (const float*)d_in[1];
    const float* rc  = (const float*)d_in[2];
    const unsigned char* mask = (const unsigned char*)d_in[3];   // jax bool_
    const int*   pc  = (const int*)d_in[4];
    const float* sW0 = (const float*)d_in[5],  *sb0 = (const float*)d_in[6];
    const float* sW1 = (const float*)d_in[7],  *sb1 = (const float*)d_in[8];
    const float* sW2 = (const float*)d_in[9],  *sb2 = (const float*)d_in[10];
    const float* hW0 = (const float*)d_in[11], *hb0 = (const float*)d_in[12];
    const float* hW1 = (const float*)d_in[13], *hb1 = (const float*)d_in[14];
    const float* hW2 = (const float*)d_in[15], *hb2 = (const float*)d_in[16];
    float* ws  = (float*)d_ws;
    float* out = (float*)d_out;
    (void)in_sizes; (void)n_in; (void)out_size; (void)ws_size;

    prep_scalar<<<dim3(1), dim3(128), 0, stream>>>(tv, tm, rc, pc,
                                                   sW0, sb0, sW1, sb1, sW2, sb2,
                                                   hW0, hb0, ws);
    prep_pack<<<dim3(72), dim3(32), 0, stream>>>(sW0, sW1, sW2, hW0, hW1, ws);

    int nblocks = (N_TRI + ROWS_PER_BLOCK - 1) / ROWS_PER_BLOCK;
    flow_main<<<dim3(nblocks), dim3(256), 0, stream>>>(tv, mask, sb0, sb1, sb2,
                                                       hb1, hW2, hb2, ws, out);
}
